// MILI_64604898066826
// MI455X (gfx1250) — compile-verified, matrix operand order checked by reading, not run on hardware
//
#include <hip/hip_runtime.h>
#include <hip/hip_bf16.h>
#include <math.h>

// Problem constants (match the reference).
#define B_   32
#define N_   512
#define IND_ 1024
#define E_   256
#define C_   16
#define DK_  64
#define NSE_ 1024

typedef __attribute__((ext_vector_type(16))) __bf16 v16bf;
typedef __attribute__((ext_vector_type(8)))  __bf16 v8bf;
typedef __attribute__((ext_vector_type(8)))  float  v8f;

__device__ __forceinline__ __bf16 f2bf(float x) {
  union { float f; unsigned u; } v; v.f = x;
  unsigned r = v.u + 0x7FFFu + ((v.u >> 16) & 1u);   // round-to-nearest-even
  unsigned short h = (unsigned short)(r >> 16);
  return __builtin_bit_cast(__bf16, h);
}

__device__ __forceinline__ v8f wmma_bf16(v16bf a, v16bf b, v8f c) {
  // D = A(16x32 bf16) x B(32x16 bf16) + C(16x16 f32)
  return __builtin_amdgcn_wmma_f32_16x16x32_bf16(false, a, false, b, (short)0, c,
                                                 false, false);
}

// 16x32 bf16 fragment from memory laid out "fragment-row contiguous":
// element [r][k] = base[r*ld + k].  Per the ISA layout (lanes 0-15: rows 0-15,
// K = {0..7,16..23}; lanes 16-31: same rows, K += 8) each lane needs exactly
// two contiguous 16-byte chunks -> two b128 loads, zero conversion ALU.
__device__ __forceinline__ v16bf load_frag_bf16(const __bf16* base, int ld, int lane) {
  const int half8 = (lane & 16) ? 8 : 0;
  const __bf16* rp = base + (lane & 15) * ld + half8;
  v8bf lo = *(const v8bf*)(rp);
  v8bf hi = *(const v8bf*)(rp + 16);
  return __builtin_shufflevector(lo, hi, 0, 1, 2, 3, 4, 5, 6, 7,
                                         8, 9, 10, 11, 12, 13, 14, 15);
}

// ---------------------------------------------------------------------------
// K0: one-time weight conversion.  liw/low -> bf16 (already W^T layout);
//     VV[l][c][e][dk] -> vvt[l][c][dk][e] bf16 (so B fragments are contiguous).
// ---------------------------------------------------------------------------
__global__ void k0_prep(const float* __restrict__ liw, const float* __restrict__ low,
                        const float* __restrict__ VV, __bf16* __restrict__ liw_bf,
                        __bf16* __restrict__ low_bf, __bf16* __restrict__ vvt) {
  const size_t i = (size_t)blockIdx.x * blockDim.x + threadIdx.x;
  const size_t stride = (size_t)gridDim.x * blockDim.x;
  for (size_t j = i; j < (size_t)E_ * IND_; j += stride) liw_bf[j] = f2bf(liw[j]);
  for (size_t j = i; j < (size_t)NSE_ * C_ * E_; j += stride) low_bf[j] = f2bf(low[j]);
  for (size_t j = i; j < (size_t)3 * C_ * DK_ * E_; j += stride) {
    const size_t e = j & (E_ - 1);
    const size_t d = (j >> 8) & (DK_ - 1);
    const size_t lc = j >> 14;                       // layer*C + c
    vvt[j] = f2bf(VV[(lc * E_ + e) * DK_ + d]);
  }
}

// ---------------------------------------------------------------------------
// K1: re0[B*N, E] = inp @ li_w^T + li_b.  One 16-row M strip per block:
// stage the f32 strip to LDS as bf16 once, then 8 waves x 2 N-tiles of WMMA.
// Writes re0 (f32) and re0b (bf16).
// ---------------------------------------------------------------------------
__global__ void k1_proj(const float* __restrict__ inp, const __bf16* __restrict__ liw_bf,
                        const float* __restrict__ lib, float* __restrict__ re0,
                        __bf16* __restrict__ re0b) {
  __shared__ __bf16 a_lds[16 * IND_];   // 32 KB
  const int tid  = threadIdx.x;
  const int lane = tid & 31;
  const int wave = tid >> 5;
  const int mt = blockIdx.x;            // 0..1023

  // Stage + convert the 16x1024 A strip (each element converted exactly once).
  const float* strip = inp + (size_t)mt * 16 * IND_;
  for (int i = tid; i < 16 * IND_; i += 256) a_lds[i] = f2bf(strip[i]);
  __syncthreads();

  const int nt0 = wave * 2;             // 2 N-tiles per wave -> all 16 tiles
  v8f acc[2] = {};
  for (int k0 = 0; k0 < IND_; k0 += 32) {
    v16bf a = load_frag_bf16(a_lds + k0, IND_, lane);
#pragma unroll
    for (int j = 0; j < 2; ++j) {
      v16bf b = load_frag_bf16(liw_bf + (size_t)(nt0 + j) * 16 * IND_ + k0, IND_, lane);
      acc[j] = wmma_bf16(a, b, acc[j]);
    }
  }
  const int m0 = mt * 16 + ((lane & 16) ? 8 : 0);
#pragma unroll
  for (int j = 0; j < 2; ++j) {
    const int n = (nt0 + j) * 16 + (lane & 15);
    const float bs = lib[n];
#pragma unroll
    for (int v = 0; v < 8; ++v) {
      const float r = acc[j][v] + bs;
      re0[(size_t)(m0 + v) * E_ + n]  = r;
      re0b[(size_t)(m0 + v) * E_ + n] = f2bf(r);
    }
  }
}

// ---------------------------------------------------------------------------
// K2: layer-0 gated attention. One workgroup (8 waves) per (b, c) pair.
//   h = tanh(re0[b] @ VV[0,c]); s = h @ WW[0,c]; masked softmax; re1 = relu(a^T re0[b])
// ---------------------------------------------------------------------------
__global__ void k2_layer0(const float* __restrict__ re0, const __bf16* __restrict__ re0b,
                          const unsigned char* __restrict__ mask,
                          const __bf16* __restrict__ vvt, const float* __restrict__ WW,
                          float* __restrict__ re1) {
  __shared__ float s_s[N_];
  __shared__ float a_s[N_];
  __shared__ float red[256];
  __shared__ float ww_s[DK_];
  const int tid  = threadIdx.x;
  const int lane = tid & 31;
  const int wave = tid >> 5;
  const int b = blockIdx.x >> 4;
  const int c = blockIdx.x & 15;

  if (tid < DK_) ww_s[tid] = WW[c * DK_ + tid];             // layer 0
  __syncthreads();

  const __bf16* vv0 = vvt + (size_t)c * DK_ * E_;           // [dk][e] bf16

  // Phase A: attention logits s[0..511] via WMMA, tanh, dot with WW.
  for (int mtile = wave; mtile < N_ / 16; mtile += 8) {
    v8f acc[4] = {};
    const __bf16* abase = re0b + (size_t)(b * N_ + mtile * 16) * E_;
    for (int k0 = 0; k0 < E_; k0 += 32) {
      v16bf a = load_frag_bf16(abase + k0, E_, lane);
#pragma unroll
      for (int t = 0; t < 4; ++t) {
        v16bf bb = load_frag_bf16(vv0 + (size_t)(t * 16) * E_ + k0, E_, lane);
        acc[t] = wmma_bf16(a, bb, acc[t]);
      }
    }
    const int nlo = lane & 15;
    float p[8];
#pragma unroll
    for (int v = 0; v < 8; ++v) {
      float sum = 0.f;
#pragma unroll
      for (int t = 0; t < 4; ++t) sum += tanhf(acc[t][v]) * ww_s[t * 16 + nlo];
      p[v] = sum;
    }
#pragma unroll
    for (int v = 0; v < 8; ++v) {
#pragma unroll
      for (int off = 8; off > 0; off >>= 1) p[v] += __shfl_xor(p[v], off, 32);
    }
    if (nlo == 0) {
      const int m0 = mtile * 16 + ((lane & 16) ? 8 : 0);
#pragma unroll
      for (int v = 0; v < 8; ++v) s_s[m0 + v] = p[v];
    }
  }
  __syncthreads();

  // Masked softmax over the 512 instances.
  const unsigned char* mrow = mask + (size_t)(b * C_ + c) * N_;
  const float NEG = -__builtin_inff();
  float x1 = mrow[tid]       ? s_s[tid]       : NEG;
  float x2 = mrow[tid + 256] ? s_s[tid + 256] : NEG;
  red[tid] = fmaxf(x1, x2);
  __syncthreads();
  for (int off = 128; off > 0; off >>= 1) {
    if (tid < off) red[tid] = fmaxf(red[tid], red[tid + off]);
    __syncthreads();
  }
  const float mx = red[0];
  __syncthreads();
  float e1 = mrow[tid]       ? __expf(x1 - mx) : 0.f;
  float e2 = mrow[tid + 256] ? __expf(x2 - mx) : 0.f;
  a_s[tid] = e1; a_s[tid + 256] = e2;
  red[tid] = e1 + e2;
  __syncthreads();
  for (int off = 128; off > 0; off >>= 1) {
    if (tid < off) red[tid] += red[tid + off];
    __syncthreads();
  }
  const float inv = 1.0f / red[0];
  a_s[tid] *= inv; a_s[tid + 256] *= inv;
  __syncthreads();

  // Phase B: re1[b,c,e] = relu( sum_m a[m] * re0[b,m,e] ).  re0 is L2-resident.
  float accv = 0.f;
  const float* rb = re0 + (size_t)b * N_ * E_ + tid;
  for (int m = 0; m < N_; ++m) accv += a_s[m] * rb[(size_t)m * E_];
  re1[(size_t)(b * C_ + c) * E_ + tid] = fmaxf(accv, 0.f);
}

// ---------------------------------------------------------------------------
// K3: layers 1 and 2, fully LDS-resident per bag. One workgroup per b.
// Emits re2b (bf16) for the final GEMM.
// ---------------------------------------------------------------------------
__global__ void k3_layers12(const float* __restrict__ re1, const __bf16* __restrict__ vvt,
                            const float* __restrict__ WW, __bf16* __restrict__ re2b) {
  __shared__ float  cur[C_ * E_];    // 16x256 current re (f32)
  __shared__ __bf16 curb[C_ * E_];   // bf16 mirror for WMMA A
  __shared__ float  nxt[C_ * E_];
  __shared__ float  s_s[C_ * C_];    // [c][m]
  __shared__ float  a_s[C_ * C_];
  const int tid  = threadIdx.x;
  const int lane = tid & 31;
  const int wave = tid >> 5;
  const int b = blockIdx.x;

  for (int i = tid; i < C_ * E_; i += 256) cur[i] = re1[(size_t)b * C_ * E_ + i];
  __syncthreads();

  for (int layer = 1; layer < 3; ++layer) {
    // bf16 mirror of cur (converted once per layer).
    for (int i = tid; i < C_ * E_; i += 256) curb[i] = f2bf(cur[i]);
    __syncthreads();

    // Each wave owns channels c = wave and wave+8.
#pragma unroll
    for (int cc = 0; cc < 2; ++cc) {
      const int c = wave + cc * 8;
      const __bf16* vvc = vvt + (size_t)(layer * C_ + c) * DK_ * E_;  // [dk][e]
      const float*  ww  = WW + (layer * C_ + c) * DK_;
      v8f acc[4] = {};
      for (int k0 = 0; k0 < E_; k0 += 32) {
        v16bf a = load_frag_bf16(curb + k0, E_, lane);   // single 16-row M tile
#pragma unroll
        for (int t = 0; t < 4; ++t) {
          v16bf bb = load_frag_bf16(vvc + (size_t)(t * 16) * E_ + k0, E_, lane);
          acc[t] = wmma_bf16(a, bb, acc[t]);
        }
      }
      const int nlo = lane & 15;
      float p[8];
#pragma unroll
      for (int v = 0; v < 8; ++v) {
        float sum = 0.f;
#pragma unroll
        for (int t = 0; t < 4; ++t) sum += tanhf(acc[t][v]) * ww[t * 16 + nlo];
        p[v] = sum;
      }
#pragma unroll
      for (int v = 0; v < 8; ++v) {
#pragma unroll
        for (int off = 8; off > 0; off >>= 1) p[v] += __shfl_xor(p[v], off, 32);
      }
      if (nlo == 0) {
        const int m0 = (lane & 16) ? 8 : 0;
#pragma unroll
        for (int v = 0; v < 8; ++v) s_s[c * C_ + m0 + v] = p[v];
      }
    }
    __syncthreads();
    // Per-channel softmax over the 16 instances (one thread per channel).
    if (tid < C_) {
      const int c = tid;
      float mx = -__builtin_inff();
      for (int m = 0; m < C_; ++m) mx = fmaxf(mx, s_s[c * C_ + m]);
      float ev[C_]; float sum = 0.f;
      for (int m = 0; m < C_; ++m) { ev[m] = __expf(s_s[c * C_ + m] - mx); sum += ev[m]; }
      const float inv = 1.0f / sum;
      for (int m = 0; m < C_; ++m) a_s[c * C_ + m] = ev[m] * inv;
    }
    __syncthreads();
    // Weighted sum + relu: nxt[c][e] = relu( sum_m a[c][m] * cur[m][e] ).
    for (int idx = tid; idx < C_ * E_; idx += 256) {
      const int c = idx >> 8;
      const int e = idx & (E_ - 1);
      float acc2 = 0.f;
#pragma unroll
      for (int m = 0; m < C_; ++m) acc2 += a_s[c * C_ + m] * cur[m * E_ + e];
      nxt[idx] = fmaxf(acc2, 0.f);
    }
    __syncthreads();
    for (int i = tid; i < C_ * E_; i += 256) cur[i] = nxt[i];
    __syncthreads();
  }
  for (int i = tid; i < C_ * E_; i += 256)
    re2b[(size_t)b * C_ * E_ + i] = f2bf(cur[i]);
}

// ---------------------------------------------------------------------------
// K4: out[32,1024] = relu( re2[32,4096] @ lo_w[1024,4096]^T + lo_b ), bf16 WMMA.
// ---------------------------------------------------------------------------
__global__ void k4_out(const __bf16* __restrict__ re2b, const __bf16* __restrict__ low_bf,
                       const float* __restrict__ lob, float* __restrict__ out) {
  const int lane = threadIdx.x & 31;
  const int wave = threadIdx.x >> 5;
  const int t  = blockIdx.x * 8 + wave;  // 0..127
  const int mt = t >> 6;                 // 2 M tiles
  const int nt = t & 63;                 // 64 N tiles
  const int K  = C_ * E_;                // 4096
  v8f acc = {};
  const __bf16* abase = re2b   + (size_t)(mt * 16) * K;
  const __bf16* bbase = low_bf + (size_t)(nt * 16) * K;
  for (int k0 = 0; k0 < K; k0 += 32) {
    v16bf a  = load_frag_bf16(abase + k0, K, lane);
    v16bf bb = load_frag_bf16(bbase + k0, K, lane);
    acc = wmma_bf16(a, bb, acc);
  }
  const int n    = nt * 16 + (lane & 15);
  const float bs = lob[n];
  const int m0   = mt * 16 + ((lane & 16) ? 8 : 0);
#pragma unroll
  for (int v = 0; v < 8; ++v)
    out[(size_t)(m0 + v) * NSE_ + n] = fmaxf(acc[v] + bs, 0.f);
}

// ---------------------------------------------------------------------------
extern "C" void kernel_launch(void* const* d_in, const int* in_sizes, int n_in,
                              void* d_out, int out_size, void* d_ws, size_t ws_size,
                              hipStream_t stream) {
  const float*         inp  = (const float*)d_in[0];
  const unsigned char* mask = (const unsigned char*)d_in[1];  // jnp bool (1 byte)
  const float*         liw  = (const float*)d_in[2];
  const float*         lib  = (const float*)d_in[3];
  const float*         VV   = (const float*)d_in[4];
  const float*         WW   = (const float*)d_in[5];
  const float*         low  = (const float*)d_in[6];
  const float*         lob  = (const float*)d_in[7];
  float* out = (float*)d_out;

  // Workspace layout (~35 MB): f32 buffers first, then bf16 buffers.
  float*  re0    = (float*)d_ws;                          // [B*N, E] f32
  float*  re1    = re0 + (size_t)B_ * N_ * E_;            // [B, C, E] f32
  __bf16* liw_bf = (__bf16*)(re1 + (size_t)B_ * C_ * E_); // [E, IND]
  __bf16* low_bf = liw_bf + (size_t)E_ * IND_;            // [NSE, C*E]
  __bf16* vvt    = low_bf + (size_t)NSE_ * C_ * E_;       // [3, C, DK, E]
  __bf16* re0b   = vvt + (size_t)3 * C_ * DK_ * E_;       // [B*N, E]
  __bf16* re2b   = re0b + (size_t)B_ * N_ * E_;           // [B, C*E]

  k0_prep    <<<2048, 256, 0, stream>>>(liw, low, VV, liw_bf, low_bf, vvt);
  k1_proj    <<<B_ * N_ / 16, 256, 0, stream>>>(inp, liw_bf, lib, re0, re0b);
  k2_layer0  <<<B_ * C_,      256, 0, stream>>>(re0, re0b, mask, vvt, WW, re1);
  k3_layers12<<<B_,           256, 0, stream>>>(re1, vvt, WW, re2b);
  k4_out     <<<(2 * 64) / 8, 256, 0, stream>>>(re2b, low_bf, lob, out);
}
